// SelfAttention_47115791237425
// MI455X (gfx1250) — matrix operand order repaired and checked by reading, not compiled, not run on hardware
//
#include <hip/hip_runtime.h>
#include <hip/hip_bf16.h>

// ---------------------------------------------------------------------------
// CDNA5 / gfx1250 self-attention block, all GEMMs on v_wmma_f32_16x16x32_bf16
// Weights pre-converted to bf16 once; all hot loops do pure bf16 b128 loads.
// ---------------------------------------------------------------------------

typedef __attribute__((ext_vector_type(16))) __bf16 v16bf;
typedef __attribute__((ext_vector_type(8)))  __bf16 v8bf;
typedef __attribute__((ext_vector_type(8)))  float  v8f;
typedef __attribute__((ext_vector_type(4)))  float  v4f;

#define CHN 512
#define CKD 64
#define NPIX 4096   // 64*64
#define MPIX 1024   // 32*32

__device__ __forceinline__ v8f wmma_bf16(v16bf a, v16bf b, v8f c) {
    // (neg_a, A, neg_b, B, c_mod, C, reuse_a, reuse_b)
    return __builtin_amdgcn_wmma_f32_16x16x32_bf16(false, a, false, b, (short)0, c, false, false);
}

// A fragment (16x32 bf16), lane holds row M=lane%16.
// Per-lane K runs: [2*kb, 2*kb+16) -> v[0..15] contiguous, kb = lane<16 ? 0 : 8.
__device__ __forceinline__ v16bf load_a_bf16(const __bf16* __restrict__ row, int kb) {
    v8bf lo = *(const v8bf*)(row + 2 * kb);
    v8bf hi = *(const v8bf*)(row + 2 * kb + 8);
    v16bf a;
#pragma unroll
    for (int i = 0; i < 8; ++i) { a[i] = lo[i]; a[8 + i] = hi[i]; }
    return a;
}

// B fragment (32x16 bf16): lane = column, per-lane K = 16 contiguous bf16.
// p points at the lane's 16 contiguous values (16B-aligned); two b128 loads.
__device__ __forceinline__ v16bf load_b_bf16(const __bf16* __restrict__ p) {
    v8bf lo = *(const v8bf*)(p);
    v8bf hi = *(const v8bf*)(p + 8);
    v16bf b;
#pragma unroll
    for (int i = 0; i < 8; ++i) { b[i] = lo[i]; b[8 + i] = hi[i]; }
    return b;
}

// ---------------------------------------------------------------------------
// Kernel 0: one-shot fp32 -> bf16 weight conversion.
// WallB[192][512] = [Wt;Wp;Wg], WoB[512][64] = Wo.   (256 KiB total)
// ---------------------------------------------------------------------------
__global__ __launch_bounds__(256) void k_cvt_w(
    const float* __restrict__ Wt, const float* __restrict__ Wp, const float* __restrict__ Wg,
    const float* __restrict__ Wo, __bf16* __restrict__ WallB, __bf16* __restrict__ WoB)
{
    const int idx = blockIdx.x * 256 + threadIdx.x;        // 0 .. 131071
    if (idx < 192 * CHN) {
        const int row = idx >> 9, col = idx & (CHN - 1);
        const float* src = (row < 64) ? Wt : ((row < 128) ? Wp : Wg);
        WallB[idx] = (__bf16)src[(row & 63) * CHN + col];
    } else {
        const int j = idx - 192 * CHN;                     // 0 .. 32767
        WoB[j] = (__bf16)Wo[j];
    }
}

// ---------------------------------------------------------------------------
// Kernel 1: fused theta/phi/g conv1x1.  Stacked GEMM [192 x 512] x [512 x 64].
// Block = 256 thr (8 waves), tile = 192 rows x 64 n.  x staged via LDS as
// bf16 [n][k] so conv B fragments are contiguous ds_load_b128.
// Outputs: thetaT[b][n][64] bf16, phi_pre[b][64][n] bf16, g_pre[b][64][n] bf16.
// ---------------------------------------------------------------------------
__global__ __launch_bounds__(256) void k_conv_qkv(
    const float* __restrict__ x, const __bf16* __restrict__ WallB,
    __bf16* __restrict__ thetaT, __bf16* __restrict__ phi_pre, __bf16* __restrict__ g_pre)
{
    const int b  = blockIdx.y;
    const int n0 = blockIdx.x * 64;
    __shared__ __align__(16) __bf16 xt[64][40];   // [n][k], padded row (80B, 16B-aligned)

    const int t = threadIdx.x, lane = t & 31, w = t >> 5;
    const int c  = w & 3;     // n column tile (16 wide)
    const int rh = w >> 2;    // row-tile parity: rowtiles rh + 2*i
    const int lm = lane & 15, hi = lane >> 4;
    const int kbA = hi ? 8 : 0, kbB = hi ? 16 : 0;

    v8f acc[6] = {{}, {}, {}, {}, {}, {}};

    for (int it = 0; it < 16; ++it) {
        const int k0 = it * 32;
        __syncthreads();
        { // stage x[b, k0..k0+32, n0..n0+64) -> xt[n][k] bf16
            const int nn = t & 63, kg = t >> 6;            // kg in 0..3, 8 k each
            const float* src = x + ((size_t)(b * CHN + k0 + kg * 8)) * NPIX + n0 + nn;
            v8bf pk;
#pragma unroll
            for (int j = 0; j < 8; ++j) pk[j] = (__bf16)src[(size_t)j * NPIX];
            *(v8bf*)(&xt[nn][kg * 8]) = pk;
        }
        __syncthreads();

        const v16bf bfrag = load_b_bf16(&xt[c * 16 + lm][kbB]);
#pragma unroll
        for (int i = 0; i < 6; ++i) {
            const int rt  = rh + 2 * i;                    // 0..11 over stacked [Wt;Wp;Wg]
            const v16bf afrag =
                load_a_bf16(WallB + (size_t)(rt * 16 + lm) * CHN + k0, kbA);
            acc[i] = wmma_bf16(afrag, bfrag, acc[i]);
        }
    }

    const int n_g = n0 + c * 16 + lm;                      // C col N = lane%16
#pragma unroll
    for (int i = 0; i < 6; ++i) {
        const int rt = rh + 2 * i;
        if (rt < 4) {                                      // theta -> transposed [n][ck]
            v8bf pk;
#pragma unroll
            for (int r = 0; r < 8; ++r) pk[r] = (__bf16)acc[i][r];  // ck = rt*16 + 8*hi + r
            *(v8bf*)(thetaT + ((size_t)(b * NPIX + n_g)) * CKD + rt * 16 + hi * 8) = pk;
        } else {                                           // phi/g pre-pool, natural [ck][n]
            __bf16* dst = (rt < 8) ? phi_pre : g_pre;
            const int ckb = (rt & 3) * 16 + hi * 8;
#pragma unroll
            for (int r = 0; r < 8; ++r)
                dst[((size_t)(b * CKD + ckb + r)) * NPIX + n_g] = (__bf16)acc[i][r];
        }
    }
}

// ---------------------------------------------------------------------------
// Kernel 2: 2x2 maxpool -> phiT[b][m][ck] and g[b][ck][m] (GEMM-ready layouts)
// ---------------------------------------------------------------------------
__global__ __launch_bounds__(256) void k_pool(
    const __bf16* __restrict__ phi_pre, const __bf16* __restrict__ g_pre,
    __bf16* __restrict__ phiT, __bf16* __restrict__ gP)
{
    const int idx = blockIdx.x * 256 + threadIdx.x;        // b*CKD*MPIX total
    const int m  = idx & (MPIX - 1);
    const int ck = (idx >> 10) & (CKD - 1);
    const int b  = idx >> 16;
    const int hm = m >> 5, wm = m & 31;
    const size_t base = ((size_t)(b * CKD + ck)) * NPIX + (size_t)(hm * 2) * 64 + wm * 2;

    float p = fmaxf(fmaxf((float)phi_pre[base],      (float)phi_pre[base + 1]),
                    fmaxf((float)phi_pre[base + 64], (float)phi_pre[base + 65]));
    float g = fmaxf(fmaxf((float)g_pre[base],        (float)g_pre[base + 1]),
                    fmaxf((float)g_pre[base + 64],   (float)g_pre[base + 65]));
    phiT[((size_t)(b * MPIX + m)) * CKD + ck] = (__bf16)p;
    gP[((size_t)(b * CKD + ck)) * MPIX + m]   = (__bf16)g;
}

// ---------------------------------------------------------------------------
// Kernel 3: flash attention.  Block = (b, 64 queries), 8 waves.
// Theta A-fragments are loop-invariant (hoisted).  Stream 16 tiles of 64 keys:
// S = thetaT x phiT (WMMA), online softmax in LDS, O += P x g (WMMA) with
// running-max rescale.  Writes oT[b][n][ck] bf16.
// ---------------------------------------------------------------------------
__global__ __launch_bounds__(256) void k_attn(
    const __bf16* __restrict__ thetaT, const __bf16* __restrict__ phiT,
    const __bf16* __restrict__ gP, __bf16* __restrict__ oT)
{
    const int b  = blockIdx.y;
    const int n0 = blockIdx.x * 64;
    __shared__ float S[64][65];
    __shared__ __align__(16) __bf16 P[64][72];             // 144B rows, 16B-aligned
    __shared__ float rowmax[64], rowsum[64], rowscale[64];

    const int t = threadIdx.x, lane = t & 31, w = t >> 5;
    const int nr = w >> 1;                                 // query row-tile 0..3
    const int half = w & 1;                                // m-cols / ck-cols 32*half
    const int lm = lane & 15, hi = lane >> 4;
    const int kbA = hi ? 8 : 0, kbB = hi ? 16 : 0;

    if (t < 64) { rowmax[t] = -3.0e38f; rowsum[t] = 0.0f; }
    v8f o0 = {}, o1 = {};
    const __bf16* thRow = thetaT + ((size_t)(b * NPIX + n0 + nr * 16 + lm)) * CKD;
    const v16bf aTh0 = load_a_bf16(thRow, kbA);            // K = ck 0..31  (loop-invariant)
    const v16bf aTh1 = load_a_bf16(thRow + 32, kbA);       // K = ck 32..63
    __syncthreads();

    for (int mi = 0; mi < 16; ++mi) {
        // ---- scores S[64n x 64m], this wave: 2 tiles at m = mi*64 + 32*half ----
        v8f s0 = {}, s1 = {};
        const int mBase = mi * 64 + half * 32;
#pragma unroll
        for (int ks = 0; ks < 2; ++ks) {
            const int k0 = ks * 32;
            const v16bf a = ks ? aTh1 : aTh0;
            const __bf16* p0 = phiT + ((size_t)(b * MPIX + mBase + lm))      * CKD + k0 + kbB;
            const __bf16* p1 = phiT + ((size_t)(b * MPIX + mBase + 16 + lm)) * CKD + k0 + kbB;
            s0 = wmma_bf16(a, load_b_bf16(p0), s0);
            s1 = wmma_bf16(a, load_b_bf16(p1), s1);
        }
#pragma unroll
        for (int r = 0; r < 8; ++r) {
            const int rowl = nr * 16 + r + 8 * hi;
            S[rowl][half * 32 + lm]      = s0[r];
            S[rowl][half * 32 + 16 + lm] = s1[r];
        }
        __syncthreads();

        // ---- online softmax: one thread per query row ----
        if (t < 64) {
            const float mold = rowmax[t];
            float tm = S[t][0];
            for (int j = 1; j < 64; ++j) tm = fmaxf(tm, S[t][j]);
            const float mnew = fmaxf(mold, tm);
            const float sc = __expf(mold - mnew);
            float ssum = 0.0f;
            for (int j = 0; j < 64; ++j) {
                const float p = __expf(S[t][j] - mnew);
                ssum += p;
                P[t][j] = (__bf16)p;
            }
            rowsum[t] = rowsum[t] * sc + ssum;
            rowmax[t] = mnew;
            rowscale[t] = sc;
        }
        __syncthreads();

        // ---- rescale O, then O += P x g : this wave ck cols 32*half ----
#pragma unroll
        for (int r = 0; r < 8; ++r) {
            const float sc = rowscale[nr * 16 + r + 8 * hi];
            o0[r] *= sc; o1[r] *= sc;
        }
#pragma unroll
        for (int ks = 0; ks < 2; ++ks) {
            const int k0 = ks * 32;                        // K = key index within tile
            const v16bf a = load_a_bf16(&P[nr * 16 + lm][k0], kbA);
            const int ck0 = half * 32;
            const __bf16* g0 = gP + ((size_t)(b * CKD + ck0 + lm))      * MPIX + mi * 64 + k0 + kbB;
            const __bf16* g1 = gP + ((size_t)(b * CKD + ck0 + 16 + lm)) * MPIX + mi * 64 + k0 + kbB;
            o0 = wmma_bf16(a, load_b_bf16(g0), o0);
            o1 = wmma_bf16(a, load_b_bf16(g1), o1);
        }
        __syncthreads();
    }

    // ---- normalize and store oT[b][n][ck] ----
#pragma unroll
    for (int r = 0; r < 8; ++r) {
        const int rowl = nr * 16 + r + 8 * hi;
        const float inv = 1.0f / rowsum[rowl];
        const size_t base = ((size_t)(b * NPIX + n0 + rowl)) * CKD + half * 32 + lm;
        oT[base]      = (__bf16)(o0[r] * inv);
        oT[base + 16] = (__bf16)(o1[r] * inv);
    }
}

// ---------------------------------------------------------------------------
// Kernel 4: out = gamma * (Wo @ o) + x.   Block tile [64 co x 128 n], 8 waves.
// ---------------------------------------------------------------------------
__global__ __launch_bounds__(256) void k_out(
    const float* __restrict__ x, const __bf16* __restrict__ WoB,
    const __bf16* __restrict__ oT, const float* __restrict__ gamma_p,
    float* __restrict__ out)
{
    const int b   = blockIdx.z;
    const int co0 = blockIdx.y * 64;
    const int n0  = blockIdx.x * 128;
    const int t = threadIdx.x, lane = t & 31, w = t >> 5;
    const int lm = lane & 15, hi = lane >> 4;
    const int kbA = hi ? 8 : 0, kbB = hi ? 16 : 0;
    const int n_g = n0 + w * 16 + lm;                      // wave owns one 16-wide n tile

    v8f acc[4] = {{}, {}, {}, {}};
#pragma unroll
    for (int ks = 0; ks < 2; ++ks) {
        const int k0 = ks * 32;
        const v16bf bfrag = load_b_bf16(oT + ((size_t)(b * NPIX + n_g)) * CKD + k0 + kbB);
#pragma unroll
        for (int ct = 0; ct < 4; ++ct) {
            const int row = co0 + ct * 16 + lm;
            const v16bf a = load_a_bf16(WoB + (size_t)row * CKD + k0, kbA);
            acc[ct] = wmma_bf16(a, bfrag, acc[ct]);
        }
    }
    const float gamma = *gamma_p;
#pragma unroll
    for (int ct = 0; ct < 4; ++ct) {
#pragma unroll
        for (int r = 0; r < 8; ++r) {
            const int co = co0 + ct * 16 + r + 8 * hi;
            const size_t idx = ((size_t)(b * CHN + co)) * NPIX + n_g;
            out[idx] = gamma * acc[ct][r] + x[idx];
        }
    }
}

// ---------------------------------------------------------------------------
// Launch
// ---------------------------------------------------------------------------
extern "C" void kernel_launch(void* const* d_in, const int* in_sizes, int n_in,
                              void* d_out, int out_size, void* d_ws, size_t ws_size,
                              hipStream_t stream) {
    (void)in_sizes; (void)n_in; (void)out_size; (void)ws_size;
    const float* x     = (const float*)d_in[0];
    const float* Wt    = (const float*)d_in[1];
    const float* Wp    = (const float*)d_in[2];
    const float* Wg    = (const float*)d_in[3];
    const float* Wo    = (const float*)d_in[4];
    const float* gamma = (const float*)d_in[5];
    float* out = (float*)d_out;

    char* ws = (char*)d_ws;
    const size_t SZ_NCK = (size_t)16 * NPIX * CKD * sizeof(__bf16);   // 8 MiB
    const size_t SZ_MCK = (size_t)16 * MPIX * CKD * sizeof(__bf16);   // 2 MiB
    __bf16* thetaT  = (__bf16*)(ws);
    __bf16* phi_pre = (__bf16*)(ws + SZ_NCK);
    __bf16* g_pre   = (__bf16*)(ws + 2 * SZ_NCK);
    __bf16* phiT    = (__bf16*)(ws + 3 * SZ_NCK);
    __bf16* gP      = (__bf16*)(ws + 3 * SZ_NCK + SZ_MCK);
    __bf16* oT      = (__bf16*)(ws + 3 * SZ_NCK + 2 * SZ_MCK);
    __bf16* WallB   = (__bf16*)(ws + 4 * SZ_NCK + 2 * SZ_MCK);
    __bf16* WoB     = WallB + (size_t)192 * CHN;

    k_cvt_w<<<dim3(512), 256, 0, stream>>>(Wt, Wp, Wg, Wo, WallB, WoB);
    k_conv_qkv<<<dim3(64, 16), 256, 0, stream>>>(x, WallB, thetaT, phi_pre, g_pre);
    k_pool<<<dim3(4096), 256, 0, stream>>>(phi_pre, g_pre, phiT, gP);
    k_attn<<<dim3(64, 16), 256, 0, stream>>>(thetaT, phiT, gP, oT);
    k_out<<<dim3(32, 8, 16), 256, 0, stream>>>(x, WoB, oT, gamma, out);
}
